// Attention_68298569941449
// MI455X (gfx1250) — matrix operand order repaired and checked by reading, not compiled, not run on hardware
//
#include <hip/hip_runtime.h>

// ---------------------------------------------------------------------------
// Shapes (compile-time constants from the reference)
// ---------------------------------------------------------------------------
#define DIM       1024
#define HEADS     16
#define HEAD_DIM  64
#define NPATCH    1024
#define BATCH     4
#define SCALE     (1.0f / 64.0f)

typedef __bf16         v16bf __attribute__((ext_vector_type(16)));
typedef __bf16         v8bf  __attribute__((ext_vector_type(8)));
typedef float          v8f   __attribute__((ext_vector_type(8)));

// ---------------------------------------------------------------------------
// Helpers
// ---------------------------------------------------------------------------
__device__ __forceinline__ unsigned short f32_to_bf16(float f) {
    unsigned int u = __float_as_uint(f);
    unsigned int r = u + 0x7FFFu + ((u >> 16) & 1u);   // round-to-nearest-even
    return (unsigned short)(r >> 16);
}

__device__ __forceinline__ v8bf ld8(const unsigned short* p) {
    return *reinterpret_cast<const v8bf*>(p);
}
__device__ __forceinline__ v16bf cat8(v8bf lo, v8bf hi) {
    return __builtin_shufflevector(lo, hi, 0,1,2,3,4,5,6,7,8,9,10,11,12,13,14,15);
}

// A-matrix 16x32 bf16 tile (ISA 7.12.2): lane<16 -> row=lane, K {0..7,16..23};
// lane>=16 -> row=lane-16, K {8..15,24..31}.  M row-major, leading dim ld.
__device__ __forceinline__ v16bf load_A(const unsigned short* M, int ld,
                                        int row0, int k0, int lane) {
    int r  = lane & 15;
    int ko = (lane >> 4) << 3;               // 0 or 8
    const unsigned short* p = M + (size_t)(row0 + r) * ld + k0 + ko;
    return cat8(ld8(p), ld8(p + 16));
}

// B-matrix 32x16 bf16 tile where B = W^T: column (col0+lane%16) of B is row
// (col0+lane%16) of W.  lane<16 -> K {k0..k0+7,k0+16..k0+23};
// lane>=16 -> K {k0+8..k0+15,k0+24..k0+31}.
__device__ __forceinline__ v16bf load_Bt(const unsigned short* M, int ld,
                                         int col0, int k0, int lane) {
    int c  = lane & 15;
    int ko = (lane >> 4) << 3;               // 0 or 8
    const unsigned short* p = M + (size_t)(col0 + c) * ld + k0 + ko;
    return cat8(ld8(p), ld8(p + 16));
}

__device__ __forceinline__ v8f wmma_bf16(v16bf a, v16bf b, v8f c) {
    return __builtin_amdgcn_wmma_f32_16x16x32_bf16(
        /*neg_a=*/false, a, /*neg_b=*/false, b,
        /*c_mod=*/(short)0, c, /*reuse_a=*/false, /*reuse_b=*/false);
}

// ---------------------------------------------------------------------------
// Kernel 0: fp32 -> bf16 conversion (grid-stride)
// ---------------------------------------------------------------------------
__global__ void cvt_bf16_kernel(const float* __restrict__ src,
                                unsigned short* __restrict__ dst, int n) {
    for (int i = blockIdx.x * blockDim.x + threadIdx.x; i < n;
         i += gridDim.x * blockDim.x) {
        dst[i] = f32_to_bf16(src[i]);
    }
}

// ---------------------------------------------------------------------------
// Kernel 1: projection GEMM  O = X @ W^T  (bf16 in, fp32 acc, bf16 out)
// X:[4096,1024]  W:[1024,1024]  O:[4096,1024].
// Register-blocked 64x32 per wave = 4x2 WMMA tiles (8 v8f accumulators,
// 64 VGPRs).  Live set ~130 VGPRs -> no spills (round-2's 4x4 blocking made
// the allocator spill accumulators to scratch inside the K-loop).
// __launch_bounds__(256, 2) budgets registers for 2 waves/SIMD.
// Per K=32 step: 12 b128 loads feed 8 WMMAs; only one B tile live at a time.
// ---------------------------------------------------------------------------
__global__ void __launch_bounds__(256, 2)
proj_wmma_kernel(const unsigned short* __restrict__ X,
                 const unsigned short* __restrict__ Wq,
                 const unsigned short* __restrict__ Wk,
                 unsigned short* __restrict__ Qo,
                 unsigned short* __restrict__ Ko) {
    int lane = threadIdx.x & 31;
    int warp = threadIdx.x >> 5;
    int gw   = blockIdx.x * 8 + warp;        // 0..4095
    int jt    = gw & 31;                     // 32 column blocks of 32
    int it    = (gw >> 5) & 63;              // 64 row blocks of 64
    int which = gw >> 11;                    // 0 = Q, 1 = K
    const unsigned short* W = which ? Wk : Wq;
    unsigned short*       O = which ? Ko : Qo;
    int i0 = it * 64, j0 = jt * 32;

    v8f acc[4][2] = {};
    for (int k0 = 0; k0 < DIM; k0 += 32) {
        v16bf a[4];
#pragma unroll
        for (int i = 0; i < 4; ++i) a[i] = load_A(X, DIM, i0 + 16 * i, k0, lane);
        if (k0 + 32 < DIM) {                 // prefetch next K-line
            __builtin_prefetch(X + (size_t)(i0 + (lane & 15)) * DIM + k0 + 32, 0, 3);
            __builtin_prefetch(W + (size_t)(j0 + (lane & 15)) * DIM + k0 + 32, 0, 3);
        }
#pragma unroll
        for (int j = 0; j < 2; ++j) {
            v16bf bt = load_Bt(W, DIM, j0 + 16 * j, k0, lane);
#pragma unroll
            for (int i = 0; i < 4; ++i)
                acc[i][j] = wmma_bf16(a[i], bt, acc[i][j]);
        }
    }

    // C layout: VGPR r = rows {r, r+8}, lane%16 = column
    int c0   = lane & 15;
    int rsel = (lane >> 4) << 3;
#pragma unroll
    for (int i = 0; i < 4; ++i) {
#pragma unroll
        for (int j = 0; j < 2; ++j) {
            int colg  = j0 + 16 * j + c0;
            int rbase = i0 + 16 * i + rsel;
#pragma unroll
            for (int r = 0; r < 8; ++r)
                O[(size_t)(rbase + r) * DIM + colg] = f32_to_bf16(acc[i][j][r]);
        }
    }
}

// ---------------------------------------------------------------------------
// Kernel 2: fused scores + softmax + shaped-attention combine.
// One 512-thread block per (b, h, 16-row strip).  Wave w holds score columns
// [w*64, w*64+64) of the strip in 4 WMMA accumulators; softmax statistics are
// reduced in-register (shfl within 16-lane groups) + tiny LDS cross-wave.
// Output hits HBM exactly once per element (the 256 MB roofline floor).
// ---------------------------------------------------------------------------
__global__ void attn_fused_kernel(const unsigned short* __restrict__ Qb,
                                  const unsigned short* __restrict__ Kb,
                                  const int*   __restrict__ mask,
                                  const float* __restrict__ g1p,
                                  const float* __restrict__ g2p,
                                  const float* __restrict__ g3p,
                                  float* __restrict__ out) {
    __shared__ float nzcol[NPATCH];      // nonzero flags per column
    __shared__ float wsum[16];           // per-wave partial nnz
    __shared__ float pmax[16 * 16];      // [row][wave] partial max
    __shared__ float psum[16 * 16];      // [row][wave] partial exp-sum

    int lane = threadIdx.x & 31;
    int w    = threadIdx.x >> 5;         // wave id 0..15
    int n0   = blockIdx.x * 16;          // row strip
    int h    = blockIdx.y;
    int b    = blockIdx.z;

    // ---- phase A: nonzero flags + partial nnz count -----------------------
    float myc = 0.f;
    for (int i = threadIdx.x; i < NPATCH; i += 512) {
        float f = (mask[b * NPATCH + i] == 0) ? 1.f : 0.f;
        nzcol[i] = f;
        myc += f;
    }
#pragma unroll
    for (int m = 16; m; m >>= 1) myc += __shfl_xor(myc, m, 32);
    if (lane == 0) wsum[w] = myc;

    // ---- phase B: WMMA scores (stay in VGPRs) -----------------------------
    const unsigned short* Qh = Qb + (size_t)b * NPATCH * DIM + h * HEAD_DIM;
    const unsigned short* Kh = Kb + (size_t)b * NPATCH * DIM + h * HEAD_DIM;

    v16bf a0 = load_A(Qh, DIM, n0, 0, lane);
    v16bf a1 = load_A(Qh, DIM, n0, 32, lane);

    v8f acc[4];
#pragma unroll
    for (int t = 0; t < 4; ++t) {
        int m0 = w * 64 + t * 16;
        v16bf b0 = load_Bt(Kh, DIM, m0, 0, lane);
        v16bf b1 = load_Bt(Kh, DIM, m0, 32, lane);
        v8f c = {};
        c = wmma_bf16(a0, b0, c);
        c = wmma_bf16(a1, b1, c);
        acc[t] = c * SCALE;
    }

    // ---- phase C: per-row max (this wave's 64 columns) --------------------
    // C layout: element r of acc = row (8*(lane>>4)+r), col = t*16 + lane%16.
    int rb = (lane >> 4) << 3;
    float rmax[8];
#pragma unroll
    for (int r = 0; r < 8; ++r) {
        float v = fmaxf(fmaxf(acc[0][r], acc[1][r]),
                        fmaxf(acc[2][r], acc[3][r]));
#pragma unroll
        for (int m = 8; m; m >>= 1) v = fmaxf(v, __shfl_xor(v, m, 32));
        rmax[r] = v;
    }
    if ((lane & 15) == 0) {
#pragma unroll
        for (int r = 0; r < 8; ++r) pmax[(rb + r) * 16 + w] = rmax[r];
    }
    __syncthreads();

    float gmax[8];
#pragma unroll
    for (int r = 0; r < 8; ++r) {
        float v = -3.4e38f;
        for (int ww = 0; ww < 16; ++ww) v = fmaxf(v, pmax[(rb + r) * 16 + ww]);
        gmax[r] = v;
    }

    // ---- phase D: exp + per-row partial sums ------------------------------
    float rsum[8];
#pragma unroll
    for (int r = 0; r < 8; ++r) rsum[r] = 0.f;
#pragma unroll
    for (int t = 0; t < 4; ++t) {
#pragma unroll
        for (int r = 0; r < 8; ++r) {
            float e = __expf(acc[t][r] - gmax[r]);
            acc[t][r] = e;
            rsum[r] += e;
        }
    }
#pragma unroll
    for (int r = 0; r < 8; ++r) {
        float v = rsum[r];
#pragma unroll
        for (int m = 8; m; m >>= 1) v += __shfl_xor(v, m, 32);
        rsum[r] = v;
    }
    if ((lane & 15) == 0) {
#pragma unroll
        for (int r = 0; r < 8; ++r) psum[(rb + r) * 16 + w] = rsum[r];
    }
    __syncthreads();

    float ginv[8];
#pragma unroll
    for (int r = 0; r < 8; ++r) {
        float v = 0.f;
        for (int ww = 0; ww < 16; ++ww) v += psum[(rb + r) * 16 + ww];
        ginv[r] = 1.0f / v;
    }

    float nnz = 0.f;
#pragma unroll
    for (int i = 0; i < 16; ++i) nnz += wsum[i];
    float invnnz = 1.0f / nnz;
    float g1 = *g1p, g2 = *g2p, g3 = *g3p;

    // ---- phase E: shaped combine + single write of the output -------------
    size_t obase = (((size_t)(b * HEADS + h)) * NPATCH + n0) * NPATCH;
    int colb = w * 64 + (lane & 15);
#pragma unroll
    for (int t = 0; t < 4; ++t) {
        int col = colb + t * 16;
        float nzm = nzcol[col];
#pragma unroll
        for (int r = 0; r < 8; ++r) {
            int row  = rb + r;          // local 0..15
            int rowg = n0 + row;
            float nzn = nzcol[rowg];
            float val = g2 * acc[t][r] * ginv[r] - g3 * nzn * nzm * invnnz;
            if (col == rowg) val += g1 * nzn;
            out[obase + (size_t)row * NPATCH + col] = val;
        }
    }
}

// ---------------------------------------------------------------------------
// Host launcher
// ---------------------------------------------------------------------------
extern "C" void kernel_launch(void* const* d_in, const int* in_sizes, int n_in,
                              void* d_out, int out_size, void* d_ws, size_t ws_size,
                              hipStream_t stream) {
    const float* hs   = (const float*)d_in[0];   // [B, N, DIM]
    const int*   mask = (const int*)  d_in[1];   // [B, N]
    const float* Wq   = (const float*)d_in[2];   // [DIM, DIM]
    const float* Wk   = (const float*)d_in[3];
    const float* g1   = (const float*)d_in[4];
    const float* g2   = (const float*)d_in[5];
    const float* g3   = (const float*)d_in[6];
    float* out = (float*)d_out;

    // workspace layout (bf16 = ushort), ~28 MB total
    char* ws = (char*)d_ws;
    unsigned short* Xbf  = (unsigned short*)(ws);                    //  8 MB
    unsigned short* Wqbf = (unsigned short*)(ws + (8u  << 20));      //  2 MB
    unsigned short* Wkbf = (unsigned short*)(ws + (10u << 20));      //  2 MB
    unsigned short* Qbf  = (unsigned short*)(ws + (12u << 20));      //  8 MB
    unsigned short* Kbf  = (unsigned short*)(ws + (20u << 20));      //  8 MB

    // k0: fp32 -> bf16
    cvt_bf16_kernel<<<2048, 256, 0, stream>>>(hs, Xbf, BATCH * NPATCH * DIM);
    cvt_bf16_kernel<<<1024, 256, 0, stream>>>(Wq, Wqbf, DIM * DIM);
    cvt_bf16_kernel<<<1024, 256, 0, stream>>>(Wk, Wkbf, DIM * DIM);

    // k1: Q/K projection GEMMs, 64x32 per wave (4096 waves, 8 per block)
    proj_wmma_kernel<<<512, 256, 0, stream>>>(Xbf, Wqbf, Wkbf, Qbf, Kbf);

    // k2: fused scores + softmax + shaped attention
    dim3 grid(NPATCH / 16, HEADS, BATCH);
    attn_fused_kernel<<<grid, 512, 0, stream>>>(Qbf, Kbf, mask, g1, g2, g3, out);
}
